// ReWanSlidingSelfAttention_78967268704666
// MI455X (gfx1250) — matrix operand order repaired and checked by reading, not compiled, not run on hardware
//
#include <hip/hip_runtime.h>
#include <stdint.h>

#define DEVI __device__ __forceinline__

typedef _Float16 v8h  __attribute__((ext_vector_type(8)));
typedef _Float16 v16h __attribute__((ext_vector_type(16)));
typedef float    v8f  __attribute__((ext_vector_type(8)));

constexpr int S_TOK = 5376;
constexpr int DIM   = 1536;
constexpr int NH    = 12;
constexpr int HD    = 128;
constexpr int IMG   = 256;   // grid_h * grid_w
constexpr int NF    = 21;    // S_TOK / IMG
constexpr int WIN   = 15;
constexpr int KVLEN = WIN * IMG; // 3840 contiguous kv tokens per frame window

DEVI v8f vzero() {
  v8f z;
#pragma unroll
  for (int i = 0; i < 8; ++i) z[i] = 0.0f;
  return z;
}

// Build a 16xf16 WMMA fragment from two contiguous 8xf16 chunks (ISA 7.12.2:
// A-frag chunks are K and K+16 apart; B-frag chunks are also K and K+16 apart).
DEVI v16h make_frag(const _Float16* p, int delta_halves) {
  const v8h lo = *(const v8h*)(p);
  const v8h hi = *(const v8h*)(p + delta_halves);
  v16h r;
#pragma unroll
  for (int i = 0; i < 8; ++i) { r[i] = lo[i]; r[i + 8] = hi[i]; }
  return r;
}

DEVI v8f wmma_f16(v16h a, v16h b, v8f c) {
  return __builtin_amdgcn_wmma_f32_16x16x32_f16(false, a, false, b,
                                                (short)0, c, false, false);
}

// CDNA5 async global->LDS copy (16B per lane), tracked with ASYNCcnt.
// Low 32 bits of a flat LDS pointer are the LDS byte address (aperture rule).
DEVI void async_g2l_b128(const void* gptr, void* lptr) {
  uint32_t lds = (uint32_t)(uintptr_t)lptr;
  asm volatile("global_load_async_to_lds_b128 %0, %1, off"
               :: "v"(lds), "v"(gptr)
               : "memory");
}
DEVI void wait_async0() {
  asm volatile("s_wait_asynccnt 0x0" ::: "memory");
}

// ---------------------------------------------------------------- convert
__global__ __launch_bounds__(256) void cvt_f32_to_f16_kernel(
    const float* __restrict__ in, _Float16* __restrict__ out, int n) {
  int i = blockIdx.x * blockDim.x + threadIdx.x;
  if (i < n) out[i] = (_Float16)in[i];
}

// ---------------------------------------------------------------- GEMM (NT)
// C[M,N] = A[M,K] * B[N,K]^T + bias.  A,B f16 row-major (K contiguous).
// LDS double-buffered; tiles arrive via async global->LDS DMA while WMMAs
// consume the previous tile.
template <bool OUT_F32>
__global__ __launch_bounds__(256) void gemm_nt_wmma_kernel(
    const _Float16* __restrict__ A, const _Float16* __restrict__ B,
    const float* __restrict__ bias, void* __restrict__ Cout,
    int M, int N, int K) {
  constexpr int BM = 128, BN = 128, BK = 32;
  constexpr int LDT = BK + 8; // 40 halves -> 80B row stride (16B aligned)
  __shared__ __align__(16) _Float16 sA[2][BM * LDT];
  __shared__ __align__(16) _Float16 sB[2][BN * LDT];

  const int tid  = threadIdx.x;
  const int lane = tid & 31;
  const int wave = tid >> 5;
  const int l16  = lane & 15;
  const int lhf  = lane >> 4;
  const int wm   = wave & 3;   // 4 waves along M (32 rows each)
  const int wn   = wave >> 2;  // 2 waves along N (64 cols each)
  const int bm   = blockIdx.y * BM;
  const int bn   = blockIdx.x * BN;

  v8f acc[2][4];
#pragma unroll
  for (int t = 0; t < 2; ++t)
#pragma unroll
    for (int n = 0; n < 4; ++n) acc[t][n] = vzero();

  // Async-stage one 128x32 A tile + 128x32 B tile into buffer `buf`.
  auto stage = [&](int buf, int k0) {
#pragma unroll
    for (int c = 0; c < 2; ++c) {
      int e   = (c * 256 + tid) * 8;
      int r   = e >> 5;        // /32
      int col = e & 31;
      async_g2l_b128(&A[(size_t)(bm + r) * K + k0 + col],
                     &sA[buf][r * LDT + col]);
      async_g2l_b128(&B[(size_t)(bn + r) * K + k0 + col],
                     &sB[buf][r * LDT + col]);
    }
  };

  stage(0, 0);
  int cur = 0;
  for (int k0 = 0; k0 < K; k0 += BK) {
    wait_async0();      // my wave's async copies (tile for `cur`) landed
    __syncthreads();    // all waves' copies landed + prev tile fully consumed
    if (k0 + BK < K) stage(cur ^ 1, k0 + BK);  // prefetch next tile

    v16h af[2], bf[4];
#pragma unroll
    for (int t = 0; t < 2; ++t) {
      int row = wm * 32 + t * 16 + l16;
      af[t] = make_frag(&sA[cur][row * LDT + lhf * 8], 16);
    }
#pragma unroll
    for (int n = 0; n < 4; ++n) {
      int col = wn * 64 + n * 16 + l16;
      bf[n] = make_frag(&sB[cur][col * LDT + lhf * 8], 16);
    }
#pragma unroll
    for (int t = 0; t < 2; ++t)
#pragma unroll
      for (int n = 0; n < 4; ++n)
        acc[t][n] = wmma_f16(af[t], bf[n], acc[t][n]);
    cur ^= 1;
  }

  // Epilogue: C/D layout — VGPR r, lane -> M = 16t + 8*(lane/16) + r, N = lane%16.
#pragma unroll
  for (int t = 0; t < 2; ++t) {
#pragma unroll
    for (int n = 0; n < 4; ++n) {
      int gn = bn + wn * 64 + n * 16 + l16;
      float bv = bias[gn];
#pragma unroll
      for (int r = 0; r < 8; ++r) {
        int gm = bm + wm * 32 + t * 16 + lhf * 8 + r;
        float v = acc[t][n][r] + bv;
        if (OUT_F32)
          ((float*)Cout)[(size_t)gm * N + gn] = v;
        else
          ((_Float16*)Cout)[(size_t)gm * N + gn] = (_Float16)v;
      }
    }
  }
}

// ------------------------------------------------------- RMSNorm + RoPE (q,k)
__global__ __launch_bounds__(256) void rmsnorm_rope_kernel(
    _Float16* __restrict__ q, _Float16* __restrict__ k,
    const float* __restrict__ gq, const float* __restrict__ gk,
    const float* __restrict__ freqs) {
  const int s = blockIdx.x;
  _Float16* buf  = blockIdx.y ? k : q;
  const float* g = blockIdx.y ? gk : gq;
  const int tid = threadIdx.x;
  const size_t base = (size_t)s * DIM;

  float ss = 0.0f;
  for (int i = tid; i < DIM; i += 256) {
    float v = (float)buf[base + i];
    ss += v * v;
  }
#pragma unroll
  for (int m = 16; m; m >>= 1) ss += __shfl_xor(ss, m, 32);
  __shared__ float part[8];
  if ((tid & 31) == 0) part[tid >> 5] = ss;
  __syncthreads();
  float tot = part[0] + part[1] + part[2] + part[3] +
              part[4] + part[5] + part[6] + part[7];
  const float rn = rsqrtf(tot / (float)DIM + 1e-6f);

  for (int p = tid; p < DIM / 2; p += 256) {
    int h  = p / (HD / 2);
    int j  = p % (HD / 2);
    int i0 = h * HD + 2 * j;
    float fr = freqs[(size_t)s * (HD / 2) + j];
    float c = __cosf(fr), sn = __sinf(fr);
    float e = (float)buf[base + i0]     * rn * g[i0];
    float o = (float)buf[base + i0 + 1] * rn * g[i0 + 1];
    buf[base + i0]     = (_Float16)(e * c - o * sn);
    buf[base + i0 + 1] = (_Float16)(e * sn + o * c);
  }
}

// ------------------------------------------- sliding-window flash attention
__global__ __launch_bounds__(256) void attn_sliding_wmma_kernel(
    const _Float16* __restrict__ qh, const _Float16* __restrict__ kh,
    const _Float16* __restrict__ vh, _Float16* __restrict__ oh) {
  constexpr int BKV = 32;
  constexpr int LDK = HD + 8;   // K tile row stride (halves)
  constexpr int LDV = BKV + 8;  // V^T row stride
  constexpr int LDP = BKV + 8;  // P stage row stride
  __shared__ __align__(16) _Float16 sK[BKV * LDK];      // 32 keys x 128 d
  __shared__ __align__(16) _Float16 sVT[HD * LDV];      // 128 d x 32 keys
  __shared__ __align__(16) _Float16 sP[8 * 32 * LDP];   // per-wave P (32 x 32)

  const int f    = blockIdx.x;
  const int h    = blockIdx.y;
  const int tid  = threadIdx.x;
  const int lane = tid & 31;
  const int wave = tid >> 5;
  const int l16  = lane & 15;
  const int lhf  = lane >> 4;

  int start = f - WIN / 2;
  if (start < 0) start = 0;
  if (start > NF - WIN) start = NF - WIN;
  const size_t kv0 = (size_t)start * IMG;
  const size_t q0  = (size_t)f * IMG + wave * 32;

  const _Float16 hscale = (_Float16)0.08838834764831845f; // 1/sqrt(128)

  // Preload Q fragments (2 M-tiles x 4 K-steps); fold softmax scale into Q.
  v16h qf[2][4];
#pragma unroll
  for (int t = 0; t < 2; ++t)
#pragma unroll
    for (int ks = 0; ks < 4; ++ks) {
      size_t tok = q0 + t * 16 + l16;
      v16h fr = make_frag(&qh[tok * DIM + h * HD + ks * 32 + lhf * 8], 16);
#pragma unroll
      for (int i = 0; i < 16; ++i) fr[i] = fr[i] * hscale;
      qf[t][ks] = fr;
    }

  v8f oacc[2][8];
#pragma unroll
  for (int t = 0; t < 2; ++t)
#pragma unroll
    for (int n = 0; n < 8; ++n) oacc[t][n] = vzero();
  float mrow[2][8], lrow[2][8];
#pragma unroll
  for (int t = 0; t < 2; ++t)
#pragma unroll
    for (int r = 0; r < 8; ++r) { mrow[t][r] = -1e30f; lrow[t][r] = 0.0f; }

  for (int kb = 0; kb < KVLEN; kb += BKV) {
    __syncthreads();   // previous iteration's LDS reads complete everywhere
    // K tile via CDNA5 async global->LDS DMA: 32 keys x 128 halves.
#pragma unroll
    for (int c = 0; c < 2; ++c) {
      int e   = (c * 256 + tid) * 8;
      int key = e >> 7;        // /128
      int col = e & 127;
      async_g2l_b128(&kh[(kv0 + kb + key) * DIM + h * HD + col],
                     &sK[key * LDK + col]);
    }
    // V tile staged transposed (register transpose) so PV B-fragments are
    // K-contiguous in LDS.
#pragma unroll
    for (int c = 0; c < 2; ++c) {
      int e   = c * 256 + tid;   // 512 v8h chunks
      int key = e / 16;
      int d8  = (e % 16) * 8;
      v8h v = *(const v8h*)&vh[(kv0 + kb + key) * DIM + h * HD + d8];
#pragma unroll
      for (int i = 0; i < 8; ++i) sVT[(d8 + i) * LDV + key] = v[i];
    }
    wait_async0();
    __syncthreads();

    // S = (Q*scale) K^T for 32 queries x 32 keys (2x2 tiles, d=128 in 4 steps).
    v8f sacc[2][2];
#pragma unroll
    for (int t = 0; t < 2; ++t)
#pragma unroll
      for (int n = 0; n < 2; ++n) sacc[t][n] = vzero();
#pragma unroll
    for (int ks = 0; ks < 4; ++ks) {
      v16h bf[2];
#pragma unroll
      for (int n = 0; n < 2; ++n) {
        int key  = n * 16 + l16;
        int kcol = ks * 32 + lhf * 8;
        bf[n] = make_frag(&sK[key * LDK + kcol], 16);
      }
#pragma unroll
      for (int t = 0; t < 2; ++t)
#pragma unroll
        for (int n = 0; n < 2; ++n)
          sacc[t][n] = wmma_f16(qf[t][ks], bf[n], sacc[t][n]);
    }

    // Online softmax (row stats replicated across each 16-lane group).
#pragma unroll
    for (int t = 0; t < 2; ++t) {
#pragma unroll
      for (int r = 0; r < 8; ++r) {
        float lm = fmaxf(sacc[t][0][r], sacc[t][1][r]);
#pragma unroll
        for (int m = 8; m; m >>= 1) lm = fmaxf(lm, __shfl_xor(lm, m, 32));
        float mnew  = fmaxf(mrow[t][r], lm);
        float alpha = __expf(mrow[t][r] - mnew);
        mrow[t][r] = mnew;
        float rs = 0.0f;
#pragma unroll
        for (int n = 0; n < 2; ++n) {
          float p = __expf(sacc[t][n][r] - mnew);
          sacc[t][n][r] = p;
          rs += p;
        }
#pragma unroll
        for (int m = 8; m; m >>= 1) rs += __shfl_xor(rs, m, 32);
        lrow[t][r] = lrow[t][r] * alpha + rs;
#pragma unroll
        for (int n = 0; n < 8; ++n) oacc[t][n][r] *= alpha;
      }
    }

    // Re-layout P (C layout -> A fragment layout) through LDS.
    _Float16* pw = &sP[wave * 32 * LDP];
#pragma unroll
    for (int t = 0; t < 2; ++t)
#pragma unroll
      for (int n = 0; n < 2; ++n)
#pragma unroll
        for (int r = 0; r < 8; ++r)
          pw[(t * 16 + lhf * 8 + r) * LDP + n * 16 + l16] =
              (_Float16)sacc[t][n][r];
    __syncthreads();

    // O += P V  (M=32 rows, N=128 d, K=32 keys).
    v16h pf[2];
#pragma unroll
    for (int t = 0; t < 2; ++t)
      pf[t] = make_frag(&pw[(t * 16 + l16) * LDP + lhf * 8], 16);
#pragma unroll
    for (int n = 0; n < 8; ++n) {
      int dcol = n * 16 + l16;
      v16h bf = make_frag(&sVT[dcol * LDV + lhf * 8], 16);
#pragma unroll
      for (int t = 0; t < 2; ++t)
        oacc[t][n] = wmma_f16(pf[t], bf, oacc[t][n]);
    }
  }

  // Finalize: O / l, write f16 attention output (S, DIM) layout.
#pragma unroll
  for (int t = 0; t < 2; ++t)
#pragma unroll
    for (int n = 0; n < 8; ++n)
#pragma unroll
      for (int r = 0; r < 8; ++r) {
        size_t tok = q0 + t * 16 + lhf * 8 + r;
        float v = oacc[t][n][r] / lrow[t][r];
        oh[tok * DIM + h * HD + n * 16 + l16] = (_Float16)v;
      }
}

// ---------------------------------------------------------------- launcher
extern "C" void kernel_launch(void* const* d_in, const int* in_sizes, int n_in,
                              void* d_out, int out_size, void* d_ws,
                              size_t ws_size, hipStream_t stream) {
  (void)in_sizes; (void)n_in; (void)out_size; (void)ws_size;
  const float* x     = (const float*)d_in[0];
  const float* freqs = (const float*)d_in[1];
  const float* Wq    = (const float*)d_in[2];
  const float* bq    = (const float*)d_in[3];
  const float* Wk    = (const float*)d_in[4];
  const float* bk    = (const float*)d_in[5];
  const float* Wv    = (const float*)d_in[6];
  const float* bv    = (const float*)d_in[7];
  const float* Wo    = (const float*)d_in[8];
  const float* bo    = (const float*)d_in[9];
  const float* gq    = (const float*)d_in[10];
  const float* gk    = (const float*)d_in[11];
  float* out = (float*)d_out;

  char* ws = (char*)d_ws;
  size_t off = 0;
  auto carve = [&](size_t bytes) -> void* {
    void* p = ws + off;
    off += (bytes + 255) & ~(size_t)255;
    return p;
  };
  const size_t szX = (size_t)S_TOK * DIM * sizeof(_Float16);
  const size_t szW = (size_t)DIM * DIM * sizeof(_Float16);
  _Float16* xh  = (_Float16*)carve(szX);
  _Float16* wqh = (_Float16*)carve(szW);
  _Float16* wkh = (_Float16*)carve(szW);
  _Float16* wvh = (_Float16*)carve(szW);
  _Float16* woh = (_Float16*)carve(szW);
  _Float16* qh  = (_Float16*)carve(szX);
  _Float16* kh  = (_Float16*)carve(szX);
  _Float16* vh  = (_Float16*)carve(szX);
  _Float16* ah  = (_Float16*)carve(szX);

  const int nX = S_TOK * DIM;
  const int nW = DIM * DIM;
  cvt_f32_to_f16_kernel<<<(nX + 255) / 256, 256, 0, stream>>>(x, xh, nX);
  cvt_f32_to_f16_kernel<<<(nW + 255) / 256, 256, 0, stream>>>(Wq, wqh, nW);
  cvt_f32_to_f16_kernel<<<(nW + 255) / 256, 256, 0, stream>>>(Wk, wkh, nW);
  cvt_f32_to_f16_kernel<<<(nW + 255) / 256, 256, 0, stream>>>(Wv, wvh, nW);
  cvt_f32_to_f16_kernel<<<(nW + 255) / 256, 256, 0, stream>>>(Wo, woh, nW);

  dim3 ggrid(DIM / 128, S_TOK / 128);
  gemm_nt_wmma_kernel<false><<<ggrid, 256, 0, stream>>>(xh, wqh, bq, qh,
                                                        S_TOK, DIM, DIM);
  gemm_nt_wmma_kernel<false><<<ggrid, 256, 0, stream>>>(xh, wkh, bk, kh,
                                                        S_TOK, DIM, DIM);
  gemm_nt_wmma_kernel<false><<<ggrid, 256, 0, stream>>>(xh, wvh, bv, vh,
                                                        S_TOK, DIM, DIM);

  rmsnorm_rope_kernel<<<dim3(S_TOK, 2), 256, 0, stream>>>(qh, kh, gq, gk,
                                                          freqs);

  attn_sliding_wmma_kernel<<<dim3(NF, NH), 256, 0, stream>>>(qh, kh, vh, ah);

  gemm_nt_wmma_kernel<true><<<ggrid, 256, 0, stream>>>(ah, woh, bo, out,
                                                       S_TOK, DIM, DIM);
}
